// Encoder_63316407878247
// MI455X (gfx1250) — matrix operand order repaired and checked by reading, not compiled, not run on hardware
//
#include <hip/hip_runtime.h>

// ---------------------------------------------------------------------------
// CDNA5 (gfx1250) fused transformer encoder forward.
// bf16 WMMA (V_WMMA_F32_16X16X32_BF16) for all GEMMs + flash attention.
// Async global->LDS staging (GLOBAL_LOAD_ASYNC_TO_LDS_B128 + s_wait_asynccnt)
// with an explicitly 2-unrolled double-buffered pipeline in the GEMM.
// wave32; 256-thread blocks (8 waves).
// ---------------------------------------------------------------------------

typedef __attribute__((ext_vector_type(16))) __bf16 v16bf;
typedef __attribute__((ext_vector_type(8)))  float  v8f;
typedef __attribute__((ext_vector_type(4)))  int    v4i;

#define DEV static __device__ __forceinline__

#if __has_builtin(__builtin_amdgcn_global_load_async_to_lds_b128)
#define HAVE_ASYNC_LDS 1
#else
#define HAVE_ASYNC_LDS 0
#warning "gfx1250 async-to-LDS builtin not available; falling back to sync staging"
#endif

DEV __bf16 tobf(float f) {
  union { float f; unsigned u; } x; x.f = f;
  unsigned r = (x.u + 0x7FFFu + ((x.u >> 16) & 1u)) >> 16;  // RNE
  union { unsigned short s; __bf16 b; } y; y.s = (unsigned short)r;
  return y.b;
}

union FragAB { v16bf v; uint4 q[2]; };
union Pack8  { uint4 q; __bf16 e[8]; };

DEV v8f wmma_bf16(v16bf a, v16bf b, v8f c) {
  // D = A(16x32 bf16) * B(32x16 bf16) + C(16x16 f32)
  return __builtin_amdgcn_wmma_f32_16x16x32_bf16(false, a, false, b,
                                                 (short)0, c, false, false);
}

// 16-byte global -> LDS copy, async when the async-to-LDS path is available.
DEV void async_copy16(const void* g, void* l) {
#if HAVE_ASYNC_LDS
  __builtin_amdgcn_global_load_async_to_lds_b128(
      (__attribute__((address_space(1))) v4i*)(g),
      (__attribute__((address_space(3))) v4i*)(l), 0, 0);
#else
  *reinterpret_cast<uint4*>(l) = *reinterpret_cast<const uint4*>(g);
#endif
}

DEV void wait_async_le4() {
#if HAVE_ASYNC_LDS
  asm volatile("s_wait_asynccnt 4" ::: "memory");
#endif
}
DEV void wait_async_0() {
#if HAVE_ASYNC_LDS
  asm volatile("s_wait_asynccnt 0" ::: "memory");
#endif
}

// ---------------------------------------------------------------------------
// GEMM: out[M][N] = A[M][K](bf16) @ W[N][K]^T(bf16) + bias  (epilogue variants)
// EP bit0: store f32   bit1: store bf16   bit2: relu   bit3: add residual f32
// Block tile 128x128, 8 waves of 32x64, K-step 32.
// Double-buffered async-LDS pipeline, explicitly unrolled by 2 so the buffer
// index is a compile-time constant (constant LDS offsets, full WMMA unroll).
// ---------------------------------------------------------------------------
template <int EP>
__global__ __launch_bounds__(256) void gemm_kernel(
    const __bf16* __restrict__ A, const __bf16* __restrict__ W,
    const float* __restrict__ bias, const float* __restrict__ res,
    float* __restrict__ outF, __bf16* __restrict__ outB,
    int M, int N, int K, float scale) {
  __shared__ __bf16 Ash[2][128 * 32];
  __shared__ __bf16 Bsh[2][128 * 32];
  const int tid  = threadIdx.x;
  const int lane = tid & 31, wave = tid >> 5;
  const int l16  = lane & 15, hi = lane >> 4;
  const int wr   = (wave >> 1) * 32;   // wave row offset in tile
  const int wc   = (wave & 1) * 64;    // wave col offset in tile
  const int bm   = blockIdx.y * 128, bn = blockIdx.x * 128;

  v8f acc[2][4];
#pragma unroll
  for (int i = 0; i < 2; i++)
#pragma unroll
    for (int j = 0; j < 4; j++)
#pragma unroll
      for (int r = 0; r < 8; r++) acc[i][j][r] = 0.0f;

  // issue one K-slab (128x32 A + 128x32 B): 4 async B128 copies per thread
  auto issue = [&](int buf, int k0) {
#pragma unroll
    for (int c = tid; c < 512; c += 256) {
      int row = c >> 2, cc = (c & 3) * 8;
      async_copy16(A + (size_t)(bm + row) * K + k0 + cc,
                   &Ash[buf][row * 32 + cc]);
      async_copy16(W + (size_t)(bn + row) * K + k0 + cc,
                   &Bsh[buf][row * 32 + cc]);
    }
  };

  // 8 WMMAs on one staged slab (buf is a compile-time constant at call sites)
  auto compute = [&](int buf) {
    FragAB a[2];
#pragma unroll
    for (int mt = 0; mt < 2; mt++) {
      int row = wr + mt * 16 + l16;
      a[mt].q[0] = *reinterpret_cast<const uint4*>(&Ash[buf][row * 32 + hi * 8]);
      a[mt].q[1] = *reinterpret_cast<const uint4*>(&Ash[buf][row * 32 + 16 + hi * 8]);
    }
#pragma unroll
    for (int nt = 0; nt < 4; nt++) {
      int col = wc + nt * 16 + l16;
      FragAB b;
      b.q[0] = *reinterpret_cast<const uint4*>(&Bsh[buf][col * 32 + hi * 8]);
      b.q[1] = *reinterpret_cast<const uint4*>(&Bsh[buf][col * 32 + 16 + hi * 8]);
#pragma unroll
      for (int mt = 0; mt < 2; mt++)
        acc[mt][nt] = wmma_bf16(a[mt].v, b.v, acc[mt][nt]);
    }
  };

  const int NIT = K / 32;  // even (K = 512 -> 16)
  issue(0, 0);
  for (int it = 0; it < NIT; it += 2) {
    // ---- stage A: compute buffer 0, prefetch into buffer 1 ----
    if (it + 1 < NIT) { issue(1, (it + 1) * 32); wait_async_le4(); }
    else              { wait_async_0(); }
    __syncthreads();
    compute(0);
    __syncthreads();
    // ---- stage B: compute buffer 1, prefetch into buffer 0 ----
    if (it + 1 < NIT) {
      if (it + 2 < NIT) { issue(0, (it + 2) * 32); wait_async_le4(); }
      else              { wait_async_0(); }
      __syncthreads();
      compute(1);
      __syncthreads();
    }
  }

  // epilogue (C layout: VGPR r -> row r + hi*8, col = l16)
#pragma unroll
  for (int mt = 0; mt < 2; mt++)
#pragma unroll
    for (int nt = 0; nt < 4; nt++) {
      int gn = bn + wc + nt * 16 + l16;
      float bv = bias[gn];
#pragma unroll
      for (int r = 0; r < 8; r++) {
        int gm = bm + wr + mt * 16 + r + hi * 8;
        float v = (acc[mt][nt][r] + bv) * scale;
        if (EP & 4) v = fmaxf(v, 0.0f);
        if (EP & 8) v += res[(size_t)gm * N + gn];
        if (EP & 1) outF[(size_t)gm * N + gn] = v;
        if (EP & 2) outB[(size_t)gm * N + gn] = tobf(v);
      }
    }
}

// ---------------------------------------------------------------------------
// Flash attention: one block = (b,h) x 128 queries; 8 waves of 16 queries.
// scores = Q.K^T (scale pre-folded into Q), online softmax, ctx = P.V
// K chunk staged via async-to-LDS; V chunk transposed through VGPRs.
// ---------------------------------------------------------------------------
__global__ __launch_bounds__(256) void attn_kernel(
    const __bf16* __restrict__ Q, const __bf16* __restrict__ Kb,
    const __bf16* __restrict__ Vb, const unsigned char* __restrict__ mask,
    __bf16* __restrict__ Ctx) {
  constexpr int S = 1024, D = 512, DH = 64;
  __shared__ __bf16 Ksh[32 * 64];     // [key][dh]
  __shared__ __bf16 Vsh[64 * 32];     // [dh][key] (transposed)
  __shared__ float  mk[32];
  __shared__ __bf16 Psh[8][16 * 32];  // per-wave P staging

  const int tid  = threadIdx.x, lane = tid & 31, wave = tid >> 5;
  const int l16  = lane & 15, hi = lane >> 4;
  const int bh = blockIdx.y, b = bh >> 3, h = bh & 7;
  const size_t base = ((size_t)b * S) * D + (size_t)h * DH;
  const int q0 = blockIdx.x * 128 + wave * 16;

  // Q fragments: 16x64 => two 16x32 A-frags
  FragAB qf[2];
#pragma unroll
  for (int f = 0; f < 2; f++) {
    const __bf16* qp = Q + base + (size_t)(q0 + l16) * D + f * 32;
    qf[f].q[0] = *reinterpret_cast<const uint4*>(qp + hi * 8);
    qf[f].q[1] = *reinterpret_cast<const uint4*>(qp + 16 + hi * 8);
  }

  v8f O[4];
  float mrow[8], lrow[8];
#pragma unroll
  for (int d = 0; d < 4; d++)
#pragma unroll
    for (int r = 0; r < 8; r++) O[d][r] = 0.0f;
#pragma unroll
  for (int r = 0; r < 8; r++) { mrow[r] = -1e30f; lrow[r] = 0.0f; }

  for (int kc = 0; kc < S / 32; kc++) {
    __syncthreads();
    {  // stage K chunk [32x64] async; V chunk transposed [64x32] sync
      int key = tid >> 3, cc = (tid & 7) * 8;
      async_copy16(Kb + base + (size_t)(kc * 32 + key) * D + cc,
                   &Ksh[key * 64 + cc]);
      Pack8 p;
      p.q = *reinterpret_cast<const uint4*>(Vb + base + (size_t)(kc * 32 + key) * D + cc);
#pragma unroll
      for (int j = 0; j < 8; j++) Vsh[(cc + j) * 32 + key] = p.e[j];
      if (tid < 32) {
        unsigned char mv = mask[(size_t)b * S + kc * 32 + tid];
        mk[tid] = mv ? 0.0f : -1e30f;
      }
    }
    wait_async_0();
    __syncthreads();

    // scores 16x32 = two 16x16 tiles, each 2 WMMAs over dh
    float sv[2][8];
#pragma unroll
    for (int t = 0; t < 2; t++) {
      int col = t * 16 + l16;
      v8f s;
#pragma unroll
      for (int r = 0; r < 8; r++) s[r] = 0.0f;
#pragma unroll
      for (int f = 0; f < 2; f++) {
        FragAB kf;
        kf.q[0] = *reinterpret_cast<const uint4*>(&Ksh[col * 64 + f * 32 + hi * 8]);
        kf.q[1] = *reinterpret_cast<const uint4*>(&Ksh[col * 64 + f * 32 + 16 + hi * 8]);
        s = wmma_bf16(qf[f].v, kf.v, s);
      }
      float madd = mk[col];
#pragma unroll
      for (int r = 0; r < 8; r++) sv[t][r] = s[r] + madd;
    }

    // online softmax; row = r + hi*8, reduce across 16-lane half-groups
    float p[2][8], alpha[8];
#pragma unroll
    for (int r = 0; r < 8; r++) {
      float rm = fmaxf(sv[0][r], sv[1][r]);
#pragma unroll
      for (int off = 8; off >= 1; off >>= 1)
        rm = fmaxf(rm, __shfl_xor(rm, off, 32));
      float mn = fmaxf(mrow[r], rm);
      alpha[r] = __expf(mrow[r] - mn);
      mrow[r]  = mn;
      float p0 = __expf(sv[0][r] - mn), p1 = __expf(sv[1][r] - mn);
      p[0][r] = p0; p[1][r] = p1;
      float rs = p0 + p1;
#pragma unroll
      for (int off = 8; off >= 1; off >>= 1)
        rs += __shfl_xor(rs, off, 32);
      lrow[r] = lrow[r] * alpha[r] + rs;
    }
#pragma unroll
    for (int d = 0; d < 4; d++)
#pragma unroll
      for (int r = 0; r < 8; r++) O[d][r] *= alpha[r];

    // P: C layout -> per-wave LDS -> A-fragment layout
#pragma unroll
    for (int t = 0; t < 2; t++)
#pragma unroll
      for (int r = 0; r < 8; r++)
        Psh[wave][(r + hi * 8) * 32 + t * 16 + l16] = tobf(p[t][r]);
    asm volatile("s_wait_dscnt 0" ::: "memory");

    FragAB pa;
    pa.q[0] = *reinterpret_cast<const uint4*>(&Psh[wave][l16 * 32 + hi * 8]);
    pa.q[1] = *reinterpret_cast<const uint4*>(&Psh[wave][l16 * 32 + 16 + hi * 8]);

#pragma unroll
    for (int d = 0; d < 4; d++) {
      int col = d * 16 + l16;
      FragAB vf;
      vf.q[0] = *reinterpret_cast<const uint4*>(&Vsh[col * 32 + hi * 8]);
      vf.q[1] = *reinterpret_cast<const uint4*>(&Vsh[col * 32 + 16 + hi * 8]);
      O[d] = wmma_bf16(pa.v, vf.v, O[d]);
    }
  }

  // normalize and store ctx (bf16), layout [B,S,D] with head offset
#pragma unroll
  for (int d = 0; d < 4; d++)
#pragma unroll
    for (int r = 0; r < 8; r++) {
      int row = q0 + r + hi * 8;
      float v = O[d][r] / fmaxf(lrow[r], 1e-20f);
      Ctx[base + (size_t)row * D + d * 16 + l16] = tobf(v);
    }
}

// ---------------------------------------------------------------------------
// LayerNorm over D=512; one block per row; outputs f32 and/or bf16.
// ---------------------------------------------------------------------------
__global__ __launch_bounds__(256) void ln_kernel(
    const float* __restrict__ x, const float* __restrict__ g,
    const float* __restrict__ bta, float* __restrict__ outF,
    __bf16* __restrict__ outB) {
  constexpr int D = 512;
  __shared__ float r1[8], r2[8];
  const int row = blockIdx.x, tid = threadIdx.x;
  const int lane = tid & 31, wave = tid >> 5;
  const float a  = x[(size_t)row * D + tid];
  const float bb = x[(size_t)row * D + tid + 256];
  float s1 = a + bb, s2 = a * a + bb * bb;
#pragma unroll
  for (int off = 16; off >= 1; off >>= 1) {
    s1 += __shfl_xor(s1, off, 32);
    s2 += __shfl_xor(s2, off, 32);
  }
  if (lane == 0) { r1[wave] = s1; r2[wave] = s2; }
  __syncthreads();
  float t1 = 0.0f, t2 = 0.0f;
#pragma unroll
  for (int i = 0; i < 8; i++) { t1 += r1[i]; t2 += r2[i]; }
  const float mean = t1 * (1.0f / D);
  const float var  = t2 * (1.0f / D) - mean * mean;
  const float rs   = rsqrtf(var + 1e-5f);
  float y0 = (a - mean) * rs * g[tid] + bta[tid];
  float y1 = (bb - mean) * rs * g[tid + 256] + bta[tid + 256];
  if (outF) {
    outF[(size_t)row * D + tid] = y0;
    outF[(size_t)row * D + tid + 256] = y1;
  }
  if (outB) {
    outB[(size_t)row * D + tid] = tobf(y0);
    outB[(size_t)row * D + tid + 256] = tobf(y1);
  }
}

// x = e0 + e1 + e2  -> f32 + bf16
__global__ __launch_bounds__(256) void add3_kernel(
    const float4* __restrict__ a, const float4* __restrict__ b,
    const float4* __restrict__ c, float4* __restrict__ outF,
    __bf16* __restrict__ outB) {
  const size_t i = (size_t)blockIdx.x * blockDim.x + threadIdx.x;
  float4 va = a[i], vb = b[i], vc = c[i];
  float4 r;
  r.x = va.x + vb.x + vc.x; r.y = va.y + vb.y + vc.y;
  r.z = va.z + vb.z + vc.z; r.w = va.w + vb.w + vc.w;
  outF[i] = r;
  __bf16* o = outB + i * 4;
  o[0] = tobf(r.x); o[1] = tobf(r.y); o[2] = tobf(r.z); o[3] = tobf(r.w);
}

// f32 -> bf16 weight conversion (n4 = n/4 float4 groups)
__global__ __launch_bounds__(256) void cvt_kernel(
    const float* __restrict__ in, __bf16* __restrict__ out, int n4) {
  int i = blockIdx.x * 256 + threadIdx.x;
  if (i < n4) {
    float4 v = reinterpret_cast<const float4*>(in)[i];
    __bf16* o = out + (size_t)i * 4;
    o[0] = tobf(v.x); o[1] = tobf(v.y); o[2] = tobf(v.z); o[3] = tobf(v.w);
  }
}

// ---------------------------------------------------------------------------
extern "C" void kernel_launch(void* const* d_in, const int* in_sizes, int n_in,
                              void* d_out, int out_size, void* d_ws, size_t ws_size,
                              hipStream_t stream) {
  (void)in_sizes; (void)n_in; (void)out_size; (void)ws_size;
  constexpr int B = 8, S = 1024, D = 512, L = 6;
  const int M = B * S;                       // 8192
  const size_t MD  = (size_t)M * D;          // 4,194,304
  const size_t WSZ = (size_t)L * D * D;      // 1,572,864 per weight stack

  const float* e0 = (const float*)d_in[0];
  const float* e1 = (const float*)d_in[1];
  const float* e2 = (const float*)d_in[2];
  const unsigned char* mask = (const unsigned char*)d_in[3];
  const float* wq = (const float*)d_in[4];  const float* bq = (const float*)d_in[5];
  const float* wk = (const float*)d_in[6];  const float* bk = (const float*)d_in[7];
  const float* wv = (const float*)d_in[8];  const float* bv = (const float*)d_in[9];
  const float* wo = (const float*)d_in[10]; const float* bo = (const float*)d_in[11];
  const float* g1 = (const float*)d_in[12]; const float* be1 = (const float*)d_in[13];
  const float* w1 = (const float*)d_in[14]; const float* b1 = (const float*)d_in[15];
  const float* w2 = (const float*)d_in[16]; const float* b2 = (const float*)d_in[17];
  const float* g2 = (const float*)d_in[18]; const float* be2 = (const float*)d_in[19];
  const float* gf = (const float*)d_in[20]; const float* bf = (const float*)d_in[21];

  char* ws = (char*)d_ws;
  size_t off = 0;
  auto alloc = [&](size_t bytes) -> void* {
    void* p = ws + off;
    off += (bytes + 255) & ~(size_t)255;
    return p;
  };
  float*  xf  = (float*)alloc(MD * 4);
  float*  yf  = (float*)alloc(MD * 4);
  float*  o1f = (float*)alloc(MD * 4);
  __bf16* xb   = (__bf16*)alloc(MD * 2);
  __bf16* o1b  = (__bf16*)alloc(MD * 2);
  __bf16* qb   = (__bf16*)alloc(MD * 2);
  __bf16* kbuf = (__bf16*)alloc(MD * 2);
  __bf16* vbuf = (__bf16*)alloc(MD * 2);
  __bf16* ctxb = (__bf16*)alloc(MD * 2);
  __bf16* hb   = (__bf16*)alloc(MD * 2);
  __bf16* wqb = (__bf16*)alloc(WSZ * 2);
  __bf16* wkb = (__bf16*)alloc(WSZ * 2);
  __bf16* wvb = (__bf16*)alloc(WSZ * 2);
  __bf16* wob = (__bf16*)alloc(WSZ * 2);
  __bf16* w1b = (__bf16*)alloc(WSZ * 2);
  __bf16* w2b = (__bf16*)alloc(WSZ * 2);

  const int cvtGrid = (int)(WSZ / 4 / 256);
  cvt_kernel<<<cvtGrid, 256, 0, stream>>>(wq, wqb, (int)(WSZ / 4));
  cvt_kernel<<<cvtGrid, 256, 0, stream>>>(wk, wkb, (int)(WSZ / 4));
  cvt_kernel<<<cvtGrid, 256, 0, stream>>>(wv, wvb, (int)(WSZ / 4));
  cvt_kernel<<<cvtGrid, 256, 0, stream>>>(wo, wob, (int)(WSZ / 4));
  cvt_kernel<<<cvtGrid, 256, 0, stream>>>(w1, w1b, (int)(WSZ / 4));
  cvt_kernel<<<cvtGrid, 256, 0, stream>>>(w2, w2b, (int)(WSZ / 4));

  add3_kernel<<<(int)(MD / 4 / 256), 256, 0, stream>>>(
      (const float4*)e0, (const float4*)e1, (const float4*)e2, (float4*)xf, xb);

  const dim3 gg(D / 128, M / 128);  // (4, 64)
  const dim3 ga(S / 128, B * 8);    // (8, 64) : (q-blocks, b*H)

  for (int l = 0; l < L; l++) {
    const size_t wo_ = (size_t)l * D * D;
    const size_t bo_ = (size_t)l * D;
    // q,k,v projections (1/sqrt(DH)=0.125 folded into q)
    gemm_kernel<2><<<gg, 256, 0, stream>>>(xb, wqb + wo_, bq + bo_, nullptr,
                                           nullptr, qb, M, D, D, 0.125f);
    gemm_kernel<2><<<gg, 256, 0, stream>>>(xb, wkb + wo_, bk + bo_, nullptr,
                                           nullptr, kbuf, M, D, D, 1.0f);
    gemm_kernel<2><<<gg, 256, 0, stream>>>(xb, wvb + wo_, bv + bo_, nullptr,
                                           nullptr, vbuf, M, D, D, 1.0f);
    attn_kernel<<<ga, 256, 0, stream>>>(qb, kbuf, vbuf, mask, ctxb);
    // out proj + residual(x)
    gemm_kernel<9><<<gg, 256, 0, stream>>>(ctxb, wob + wo_, bo + bo_, xf,
                                           yf, nullptr, M, D, D, 1.0f);
    ln_kernel<<<M, 256, 0, stream>>>(yf, g1 + bo_, be1 + bo_, o1f, o1b);
    // FFN
    gemm_kernel<6><<<gg, 256, 0, stream>>>(o1b, w1b + wo_, b1 + bo_, nullptr,
                                           nullptr, hb, M, D, D, 1.0f);
    gemm_kernel<9><<<gg, 256, 0, stream>>>(hb, w2b + wo_, b2 + bo_, o1f,
                                           yf, nullptr, M, D, D, 1.0f);
    ln_kernel<<<M, 256, 0, stream>>>(yf, g2 + bo_, be2 + bo_, xf, xb);
  }
  ln_kernel<<<M, 256, 0, stream>>>(xf, gf, bf, (float*)d_out, nullptr);
}
